// TwoLayerGCN_2997887173231
// MI455X (gfx1250) — matrix operand order repaired and checked by reading, not compiled, not run on hardware
//
#include <hip/hip_runtime.h>

#define N_NODES 50000
#define N_EDGES 800000

typedef __attribute__((ext_vector_type(16))) __bf16 v16bf;
typedef __attribute__((ext_vector_type(8)))  float  v8f;
typedef __attribute__((ext_vector_type(4)))  float  v4f;

// ---------- helpers ----------
__device__ __forceinline__ __bf16 f2bf(float f) {
    // round-to-nearest-even f32 -> bf16
    unsigned u = __builtin_bit_cast(unsigned, f);
    unsigned r = u + 0x7FFFu + ((u >> 16) & 1u);
    unsigned short h = (unsigned short)(r >> 16);
    return __builtin_bit_cast(__bf16, h);
}

// ---------- elementwise / graph kernels ----------
__global__ void k_zero(float* __restrict__ p, long n) {
    long i = (long)blockIdx.x * blockDim.x + threadIdx.x;
    if (i < n) p[i] = 0.f;
}

__global__ void k_degree(const int* __restrict__ src, const int* __restrict__ dst,
                         float* __restrict__ degO, float* __restrict__ degI, int ne) {
    int e = blockIdx.x * blockDim.x + threadIdx.x;
    if (e < ne) {
        atomicAdd(&degO[src[e]], 1.f);
        atomicAdd(&degI[dst[e]], 1.f);
    }
}

__global__ void k_norm_inplace(float* __restrict__ d, int n) {
    int i = blockIdx.x * blockDim.x + threadIdx.x;
    if (i < n) {
        float v = d[i];
        d[i] = (v > 0.f) ? rsqrtf(v) : 0.f;   // == rsqrt(max(d,1)) for d>0
    }
}

// Scatter-add SpMM: acc[dst[e], f] += feat[src[e], f] * norm_src[src[e]]
// One thread per (edge, feature); atomics resolve in the 192MB L2.
__global__ void k_spmm(const float* __restrict__ feat, const float* __restrict__ nsrc,
                       const int* __restrict__ src, const int* __restrict__ dst,
                       float* __restrict__ acc, int ne, int F) {
    long t = (long)blockIdx.x * blockDim.x + threadIdx.x;
    long total = (long)ne * F;
    if (t >= total) return;
    int e = (int)(t / F);
    int f = (int)(t % F);
    int s = src[e];
    int d = dst[e];
    atomicAdd(&acc[(long)d * F + f], feat[(long)s * F + f] * nsrc[s]);
}

// out[i,f] = relu(acc[i,f]*norm_dst[i] + bias[f])  (final layer epilogue)
__global__ void k_final(const float* __restrict__ acc, const float* __restrict__ nd,
                        const float* __restrict__ bias, float* __restrict__ out,
                        int n, int F) {
    int t = blockIdx.x * blockDim.x + threadIdx.x;
    if (t < n * F) {
        int i = t / F, f = t % F;
        float v = acc[t] * nd[i] + bias[f];
        out[t] = v > 0.f ? v : 0.f;
    }
}

// ---------- fragment packing ----------
// A fragment layout (16-bit A 16x32, wave32): element j of lane l holds
// k = (j&7) + 8*(l>>4) + 16*(j>>3)  -> two contiguous 8-float groups per lane.
// Packed: Ap[((tm*Ksteps + ks)*32 + lane)*16 + j], one thread per (tm,ks,lane).
// Prologue (rowscale / k-bias / relu) fused here so the GEMM loop is pure.
template <bool RS, bool AB, bool RELU>
__global__ void k_packA(const float* __restrict__ A, __bf16* __restrict__ P,
                        const float* __restrict__ rowscale,
                        const float* __restrict__ abias,
                        int M, int K, int Ksteps) {
    int t = blockIdx.x * blockDim.x + threadIdx.x;
    int tilesM = (M + 15) >> 4;
    int total = tilesM * Ksteps * 32;
    if (t >= total) return;
    int lane = t & 31;
    int frag = t >> 5;
    int ks = frag % Ksteps;
    int tm = frag / Ksteps;
    int hi = lane >> 4, l15 = lane & 15;
    int row = tm * 16 + l15;
    bool rowOK = row < M;
    const float* Arow = A + (long)row * K;
    float rs = 1.f;
    if (RS && rowOK) rs = rowscale[row];
    int base = ks * 32 + (hi << 3);        // offsets used: base+{0..7}, base+16+{0..7}

    v4f q[4] = {}; v4f bq[4] = {};
#pragma unroll
    for (int g = 0; g < 4; ++g) {
        int o = base + ((g & 1) << 2) + ((g >> 1) << 4);   // 0,4,16,20
        if (rowOK && o < K) {                               // K % 16 == 0 -> whole v4f in range
            q[g] = *(const v4f*)(Arow + o);
            if (AB) bq[g] = *(const v4f*)(abias + o);
        }
    }
    v16bf outv;
#pragma unroll
    for (int j = 0; j < 16; ++j) {
        float v = q[j >> 2][j & 3] * rs;
        if (AB) v += bq[j >> 2][j & 3];
        if (RELU) v = v > 0.f ? v : 0.f;
        outv[j] = f2bf(v);
    }
    *(v16bf*)(P + (long)t * 16) = outv;
}

// B fragment layout (16-bit B 32x16, wave32): element j of lane l holds
// k = (j&7) + 8*(l>>4) + 16*(j>>3), n = l&15.  Packed: Bp[((ks*tilesN + tn)*32 + lane)*16 + j].
// B is tiny (<=128KB) so scalar guarded loads are fine here.
__global__ void k_packB(const float* __restrict__ B, __bf16* __restrict__ P,
                        int K, int N, int Ksteps, int tilesN) {
    int t = blockIdx.x * blockDim.x + threadIdx.x;
    int total = Ksteps * tilesN * 32;
    if (t >= total) return;
    int lane = t & 31;
    int frag = t >> 5;
    int tn = frag % tilesN;
    int ks = frag / tilesN;
    int hi = lane >> 4, l15 = lane & 15;
    int col = tn * 16 + l15;
    v16bf outv;
#pragma unroll
    for (int j = 0; j < 16; ++j) {
        int kb = ks * 32 + (j & 7) + (hi << 3) + ((j >> 3) << 4);
        float v = (kb < K && col < N) ? B[(long)kb * N + col] : 0.f;
        outv[j] = f2bf(v);
    }
    *(v16bf*)(P + (long)t * 16) = outv;
}

// ---------- WMMA GEMM on packed fragments ----------
// One wave per 16x(16*NT) output strip. Per K-step: 1 A-frag load (2x b128),
// NT B-frag loads, NT WMMAs -- all unconditional, software-pipelineable.
template <int NT>
__global__ void k_wmma_gemm(const __bf16* __restrict__ Ap, const __bf16* __restrict__ Bp,
                            float* __restrict__ D,
                            const float* __restrict__ d_bias, int d_relu,
                            int M, int N, int Ksteps, int tilesN) {
    int wave = threadIdx.x >> 5;
    int lane = threadIdx.x & 31;
    int tilesM = (M + 15) >> 4;
    int ntg = tilesN / NT;                       // caller guarantees tilesN % NT == 0
    int wid = blockIdx.x * (blockDim.x >> 5) + wave;
    if (wid >= tilesM * ntg) return;             // wave-uniform: EXEC all-1 at WMMA
    int tm = wid / ntg;
    int tn0 = (wid - tm * ntg) * NT;

    const v16bf* Af = (const v16bf*)Ap + ((long)tm * Ksteps) * 32 + lane;
    const v16bf* Bf = (const v16bf*)Bp + (long)tn0 * 32 + lane;

    v8f c[NT];
#pragma unroll
    for (int t = 0; t < NT; ++t) c[t] = v8f{0.f, 0.f, 0.f, 0.f, 0.f, 0.f, 0.f, 0.f};

    for (int ks = 0; ks < Ksteps; ++ks) {
        v16bf a = Af[(long)ks * 32];
#pragma unroll
        for (int t = 0; t < NT; ++t) {
            v16bf b = Bf[((long)ks * tilesN + t) * 32];
            c[t] = __builtin_amdgcn_wmma_f32_16x16x32_bf16(
                       false, a, false, b, (short)0, c[t], false, false);
        }
    }

    // D element i of lane l: row = 16*tm + i + 8*(l>>4), col = 16*tn + (l&15)
    int hi = lane >> 4, l15 = lane & 15;
#pragma unroll
    for (int t = 0; t < NT; ++t) {
        int col = (tn0 + t) * 16 + l15;
        if (col < N) {
            float db = d_bias ? d_bias[col] : 0.f;
#pragma unroll
            for (int i = 0; i < 8; ++i) {
                int r = tm * 16 + i + (hi << 3);
                if (r < M) {
                    float v = c[t][i] + db;
                    if (d_relu) v = v > 0.f ? v : 0.f;
                    D[(long)r * N + col] = v;
                }
            }
        }
    }
}

// ---------- launcher ----------
static inline int cdiv(long a, long b) { return (int)((a + b - 1) / b); }

extern "C" void kernel_launch(void* const* d_in, const int* in_sizes, int n_in,
                              void* d_out, int out_size, void* d_ws, size_t ws_size,
                              hipStream_t stream) {
    (void)in_sizes; (void)n_in; (void)out_size; (void)ws_size;
    const float* x   = (const float*)d_in[0];
    const int*   src = (const int*)d_in[1];
    const int*   dst = (const int*)d_in[2];
    const float* W1  = (const float*)d_in[3];
    const float* b1  = (const float*)d_in[4];
    const float* W2  = (const float*)d_in[5];
    const float* b2  = (const float*)d_in[6];
    const float* W3  = (const float*)d_in[7];
    const float* b3  = (const float*)d_in[8];
    const float* W4  = (const float*)d_in[9];
    const float* b4  = (const float*)d_in[10];

    const int NN = N_NODES, NE = N_EDGES;
    float*  ws       = (float*)d_ws;
    float*  norm_src = ws;                        // NN
    float*  norm_dst = ws + NN;                   // NN
    float*  bufA     = ws + 2L * NN;              // NN*128 f32 (repartitioned per layer)
    float*  bufB     = bufA + (long)NN * 128;     // NN*512 f32 (h1)
    __bf16* Ap       = (__bf16*)(bufB + (long)NN * 512);      // 3125*16*512 bf16 = 51.2MB
    __bf16* Bp       = Ap + (long)3125 * 16 * 512;            // up to 65536 bf16 = 128KB
    // total ws use ~ 180MB

    const int T = 256;

    auto packB = [&](const float* B, int K, int N) {
        int Ksteps = (K + 31) / 32, tilesN = (N + 15) / 16;
        k_packB<<<cdiv((long)Ksteps * tilesN * 32, T), T, 0, stream>>>(B, Bp, K, N, Ksteps, tilesN);
    };
    auto gemm = [&](float* D, const float* dbias, int drelu, int M, int K, int N, int NT) {
        int Ksteps = (K + 31) / 32, tilesN = (N + 15) / 16;
        int ntg = tilesN / NT;
        long waves = (long)((M + 15) / 16) * ntg;
        if (NT == 4)
            k_wmma_gemm<4><<<cdiv(waves, 8), 8 * 32, 0, stream>>>(Ap, Bp, D, dbias, drelu, M, N, Ksteps, tilesN);
        else
            k_wmma_gemm<1><<<cdiv(waves, 8), 8 * 32, 0, stream>>>(Ap, Bp, D, dbias, drelu, M, N, Ksteps, tilesN);
    };
    auto packA_plain = [&](const float* A, int M, int K) {
        int Ksteps = (K + 31) / 32;
        long tot = (long)((M + 15) / 16) * Ksteps * 32;
        k_packA<false, false, false><<<cdiv(tot, T), T, 0, stream>>>(A, Ap, nullptr, nullptr, M, K, Ksteps);
    };
    auto packA_rs = [&](const float* A, const float* rs, int M, int K) {
        int Ksteps = (K + 31) / 32;
        long tot = (long)((M + 15) / 16) * Ksteps * 32;
        k_packA<true, false, false><<<cdiv(tot, T), T, 0, stream>>>(A, Ap, rs, nullptr, M, K, Ksteps);
    };
    auto packA_full = [&](const float* A, const float* rs, const float* ab, int M, int K) {
        int Ksteps = (K + 31) / 32;
        long tot = (long)((M + 15) / 16) * Ksteps * 32;
        k_packA<true, true, true><<<cdiv(tot, T), T, 0, stream>>>(A, Ap, rs, ab, M, K, Ksteps);
    };

    // ---- degrees -> norms (in place) ----
    k_zero<<<cdiv(2L * NN, T), T, 0, stream>>>(ws, 2L * NN);
    k_degree<<<cdiv(NE, T), T, 0, stream>>>(src, dst, norm_src, norm_dst, NE);
    k_norm_inplace<<<cdiv(2L * NN, T), T, 0, stream>>>(ws, 2 * NN);

    // ---- Layer 1 (128->512, aggregate first): h1 = relu((agg1*nd) @ W1 + b1) ----
    float* agg1 = bufA;                                    // NN x 128
    k_zero<<<cdiv((long)NN * 128, T), T, 0, stream>>>(agg1, (long)NN * 128);
    k_spmm<<<cdiv((long)NE * 128, T), T, 0, stream>>>(x, norm_src, src, dst, agg1, NE, 128);
    float* h1 = bufB;                                      // NN x 512
    packA_rs(agg1, norm_dst, NN, 128);
    packB(W1, 128, 512);
    gemm(h1, b1, 1, NN, 128, 512, 4);

    // ---- Layer 2 (512->64, weight first): t2 = h1 @ W2; aggregate ----
    float* t2   = bufA;                                    // NN x 64 (agg1 dead)
    float* agg2 = bufA + (long)NN * 64;                    // NN x 64
    packA_plain(h1, NN, 512);
    packB(W2, 512, 64);
    gemm(t2, nullptr, 0, NN, 512, 64, 4);
    k_zero<<<cdiv((long)NN * 64, T), T, 0, stream>>>(agg2, (long)NN * 64);
    k_spmm<<<cdiv((long)NE * 64, T), T, 0, stream>>>(t2, norm_src, src, dst, agg2, NE, 64);

    // ---- Layer 3 (64->16): h2 = relu(agg2*nd + b2) fused into packA ----
    float* t3   = bufA;                                    // NN x 16 (t2 dead)
    float* agg3 = bufA + (long)NN * 16;                    // NN x 16
    packA_full(agg2, norm_dst, b2, NN, 64);
    packB(W3, 64, 16);
    gemm(t3, nullptr, 0, NN, 64, 16, 1);
    k_zero<<<cdiv((long)NN * 16, T), T, 0, stream>>>(agg3, (long)NN * 16);
    k_spmm<<<cdiv((long)NE * 16, T), T, 0, stream>>>(t3, norm_src, src, dst, agg3, NE, 16);

    // ---- Layer 4 (16->2): h3 = relu(agg3*nd + b3) fused into packA; K padded to 32 ----
    float* t4   = bufA + (long)NN * 32;                    // NN x 2
    float* agg4 = bufA + (long)NN * 34;                    // NN x 2
    packA_full(agg3, norm_dst, b3, NN, 16);
    packB(W4, 16, 2);
    gemm(t4, nullptr, 0, NN, 16, 2, 1);
    k_zero<<<cdiv((long)NN * 2, T), T, 0, stream>>>(agg4, (long)NN * 2);
    k_spmm<<<cdiv((long)NE * 2, T), T, 0, stream>>>(t4, norm_src, src, dst, agg4, NE, 2);

    // ---- Final epilogue: out = relu(agg4*norm_dst + b4) ----
    k_final<<<cdiv(NN * 2, T), T, 0, stream>>>(agg4, norm_dst, b4, (float*)d_out, NN, 2);
}
